// GumbelVectorQuantizer_Parallel_83124797047429
// MI455X (gfx1250) — matrix operand order repaired and checked by reading, not compiled, not run on hardware
//
#include <hip/hip_runtime.h>
#include <hip/hip_bf16.h>
#include <math.h>

// ---- problem constants (from reference) ----
#define NB   8
#define NT   2048
#define NG   2
#define NM   1024
#define ND   256
#define NN   (NB*NT)          // 16384 rows
#define NC   (NG*ND)          // 512
#define QOFF ((size_t)NN*NC)  // 8388608 floats of quantized output

typedef __attribute__((ext_vector_type(16))) __bf16 v16bf;
typedef __attribute__((ext_vector_type(8)))  float  v8f;

// ---- LDS layout for main kernel (in floats) ----
#define XS_OFF    0        // 16*256 x tile (fp32)
#define LG_OFF    4096     // 16*1024 logits
#define SQE_OFF   20480    // 1024 ||e||^2
#define RSQ_OFF   21504    // 16 ||x||^2 per row
#define RMAX_OFF  21520    // 16 row max
#define RSUM_OFF  21536    // 16 row sumexp
#define GIDX_OFF  21552    // 16 gumbel argmax (int)
#define RED_OFF   21568    // 8 wave partials
#define SMEM_FLTS 21576
#define SMEM_BYTES (SMEM_FLTS*4)

// =====================================================================
// Kernel 1: pack embedding -> bf16 WMMA-B layout, and compute ||e||^2.
// B fragment layout (16x16x32 bf16 wmma, wave32): lane&15 = N column,
// element e = K (e<8: 8*(lane>>4)+e, else 16+8*(lane>>4)+e-8).
// packB index: ((((g*64 + ct)*8 + kt)*32 + lane)*16 + e)
// =====================================================================
__global__ void gvq_prep(const float* __restrict__ emb,
                         __bf16* __restrict__ packB,
                         float* __restrict__ sqe) {
  const int bx = blockIdx.x;          // 0..127 = g*64 + ct
  const int g  = bx >> 6;
  const int ct = bx & 63;
  const int m0 = ct * 16;
  const int tid = threadIdx.x;

  // pack: 256 threads = 8 kt * 32 lanes
  const int kt   = tid >> 5;
  const int lane = tid & 31;
  {
    size_t obase = ((((size_t)g*64 + ct)*8 + kt)*32 + lane)*16;
    #pragma unroll
    for (int e = 0; e < 16; ++e) {
      float v = emb[((size_t)(g*NM + m0 + (lane & 15)))*ND + kt*32 + (e & 8)*2 + (lane >> 4)*8 + (e & 7)];
      packB[obase + e] = (__bf16)v;
    }
  }

  // ||e||^2 : 16 rows x 16-thread teams, shuffle reduce
  const int row = tid >> 4;
  const int t   = tid & 15;
  float s = 0.f;
  const float* er = emb + ((size_t)(g*NM + m0 + row))*ND;
  #pragma unroll
  for (int j = 0; j < 16; ++j) { float v = er[t + 16*j]; s += v*v; }
  #pragma unroll
  for (int off = 8; off; off >>= 1) s += __shfl_xor(s, off, 16);
  if (t == 0) sqe[g*NM + m0 + row] = s;
}

// =====================================================================
// Kernel 2: zero the accumulators (avgp[2048] | cnts[2048] | loss[1])
// =====================================================================
__global__ void gvq_zero(float* __restrict__ p, int n) {
  int i = blockIdx.x * blockDim.x + threadIdx.x;
  if (i < n) p[i] = 0.f;
}

// =====================================================================
// Kernel 3: fused main. grid (N/16, G), 256 threads = 8 waves.
// =====================================================================
__global__ __launch_bounds__(256)
void gvq_main(const float* __restrict__ x,
              const float* __restrict__ emb,
              const float* __restrict__ gum,
              const __bf16* __restrict__ packB,
              const float* __restrict__ sqe,
              float* __restrict__ avgp,
              float* __restrict__ cnts,
              float* __restrict__ lossAcc,
              float* __restrict__ outq,
              int*   __restrict__ outinds) {
  extern __shared__ float smem[];
  float* xs      = smem + XS_OFF;
  float* lg      = smem + LG_OFF;
  float* sqeL    = smem + SQE_OFF;
  float* rowsq   = smem + RSQ_OFF;
  float* rowmax  = smem + RMAX_OFF;
  float* rowsum  = smem + RSUM_OFF;
  int*   gidx    = (int*)(smem + GIDX_OFF);
  float* red     = smem + RED_OFF;

  const int tid  = threadIdx.x;
  const int wave = tid >> 5;
  const int lane = tid & 31;
  const int n0   = blockIdx.x * 16;
  const int g    = blockIdx.y;

  // prefetch the gumbel tile we will stream in pass 2 (global_prefetch_b8)
  {
    const char* pf = (const char*)(gum + ((size_t)(n0 + (tid >> 4))*NG + g)*NM)
                     + (tid & 15) * 256;
    __builtin_prefetch(pf, 0, 1);
  }

  // ---- async global -> LDS staging of the x tile (ASYNCcnt path) ----
  // 16 rows x 256 f32 = 16KB; each thread issues 4x b128 copies.
  {
    const unsigned ldsbase = (unsigned)(size_t)xs;   // low 32 bits = LDS offset
    #pragma unroll
    for (int i = 0; i < 4; ++i) {
      const int f   = i*1024 + tid*4;                // flat f32 index in tile
      const int row = f >> 8;
      const int d   = f & 255;
      const float* gp = x + ((size_t)(n0 + row))*NC + g*ND + d;
      const unsigned ldsa = ldsbase + (unsigned)f*4u;
      asm volatile("global_load_async_to_lds_b128 %0, %1, off"
                   :: "v"(ldsa), "v"(gp) : "memory");
    }
  }

  // ---- stage sq_e into LDS while the async copies fly ----
  #pragma unroll
  for (int i = tid; i < NM; i += 256)
    sqeL[i] = sqe[g*NM + i];

  asm volatile("s_wait_asynccnt 0x0" ::: "memory");
  __syncthreads();

  // ---- ||x||^2 per row (16-thread teams) ----
  {
    const int row = tid >> 4, t = tid & 15;
    float s = 0.f;
    #pragma unroll
    for (int j = 0; j < 16; ++j) { float v = xs[row*ND + t + 16*j]; s += v*v; }
    #pragma unroll
    for (int off = 8; off; off >>= 1) s += __shfl_xor(s, off, 16);
    if (t == 0) rowsq[row] = s;
  }

  // ---- GEMM: cross = X(16x256) * E^T(256x1024), wave owns 128 columns ----
  v8f acc[8];
  #pragma unroll
  for (int c = 0; c < 8; ++c) acc[c] = (v8f){0.f,0.f,0.f,0.f,0.f,0.f,0.f,0.f};

  const int arow = lane & 15;
  const int kb   = (lane >> 4) * 8;   // A-frag K sub-base per ISA layout

  #pragma unroll
  for (int kt = 0; kt < 8; ++kt) {
    // build bf16 A fragment from LDS (ISA 16-bit A 16x32 layout)
    v16bf a;
    const float* xrow = xs + arow*ND + kt*32;
    #pragma unroll
    for (int e = 0; e < 8; ++e) a[e]     = (__bf16)xrow[kb + e];
    #pragma unroll
    for (int e = 0; e < 8; ++e) a[8 + e] = (__bf16)xrow[16 + kb + e];

    #pragma unroll
    for (int c = 0; c < 8; ++c) {
      const int ct = wave*8 + c;
      const v16bf b = *(const v16bf*)(packB +
          ((((size_t)g*64 + ct)*8 + kt)*32 + lane)*16);
      acc[c] = __builtin_amdgcn_wmma_f32_16x16x32_bf16(
                   false, a, false, b, (short)0, acc[c], false, false);
    }
  }
  __syncthreads();  // rowsq visible before logits write

  // ---- logits = 2*cross - ||e||^2 - ||x||^2 -> LDS ----
  {
    const int rbase = (lane >> 4) * 8;      // C layout: lanes16-31 are M+8
    const int ncol  = lane & 15;
    #pragma unroll
    for (int c = 0; c < 8; ++c) {
      const int m = wave*128 + c*16 + ncol;
      const float se = sqeL[m];
      #pragma unroll
      for (int r = 0; r < 8; ++r) {
        const int row = rbase + r;
        lg[row*NM + m] = 2.f*acc[c][r] - se - rowsq[row];
      }
    }
  }
  __syncthreads();

  // ---- pass 2a: per-row max / argmax / gumbel-argmax / sumexp ----
  {
    const int r = tid >> 4, t = tid & 15;
    float m1 = -3.4e38f, m2 = -3.4e38f;
    int   i1 = 0,         i2 = 0;
    const float* grow = gum + ((size_t)(n0 + r)*NG + g)*NM;
    #pragma unroll 4
    for (int j = 0; j < 64; ++j) {
      const int m = t + 16*j;
      const float l = lg[r*NM + m];
      if (l > m1) { m1 = l; i1 = m; }
      const float u  = __builtin_nontemporal_load(grow + m);  // 134MB stream, keep L2 clean
      const float gl = l + (-logf(-logf(u)));     // gumbel; argmax invariant to /tau
      if (gl > m2) { m2 = gl; i2 = m; }
    }
    #pragma unroll
    for (int off = 8; off; off >>= 1) {
      float ov = __shfl_xor(m1, off, 16); int oi = __shfl_xor(i1, off, 16);
      if (ov > m1 || (ov == m1 && oi < i1)) { m1 = ov; i1 = oi; }
      ov = __shfl_xor(m2, off, 16); oi = __shfl_xor(i2, off, 16);
      if (ov > m2 || (ov == m2 && oi < i2)) { m2 = ov; i2 = oi; }
    }
    float s = 0.f;
    #pragma unroll 4
    for (int j = 0; j < 64; ++j) s += expf(lg[r*NM + t + 16*j] - m1);
    #pragma unroll
    for (int off = 8; off; off >>= 1) s += __shfl_xor(s, off, 16);
    if (t == 0) {
      rowmax[r] = m1; rowsum[r] = s; gidx[r] = i2;
      atomicAdd(&cnts[g*NM + i1], 1.0f);
      outinds[(size_t)(n0 + r)*NG + g] = i2;
    }
  }
  __syncthreads();

  // ---- pass 2b: avg softmax probs (block partial -> global atomic) ----
  #pragma unroll
  for (int mm = tid; mm < NM; mm += 256) {
    float a = 0.f;
    #pragma unroll
    for (int r = 0; r < 16; ++r)
      a += expf(lg[r*NM + mm] - rowmax[r]) / rowsum[r];
    atomicAdd(&avgp[g*NM + mm], a);
  }

  // ---- quantized gather + commitment loss ----
  float lacc = 0.f;
  #pragma unroll
  for (int r = 0; r < 16; ++r) {
    const int mi = gidx[r];
    const float e = emb[((size_t)(g*NM + mi))*ND + tid];   // L2-resident gather
    __builtin_nontemporal_store(e, outq + ((size_t)(n0 + r))*NC + g*ND + tid);
    const float d = xs[r*ND + tid] - e;
    lacc += d*d;
  }
  #pragma unroll
  for (int off = 16; off; off >>= 1) lacc += __shfl_xor(lacc, off, 32);
  __syncthreads();
  if (lane == 0) red[wave] = lacc;
  __syncthreads();
  if (tid == 0) {
    float t = 0.f;
    #pragma unroll
    for (int i = 0; i < 8; ++i) t += red[i];
    atomicAdd(lossAcc, t);
  }
}

// =====================================================================
// Kernel 4: perplexities + loss scalar. grid = G blocks.
// =====================================================================
__global__ void gvq_finalize(const float* __restrict__ cnts,
                             const float* __restrict__ avgp,
                             const float* __restrict__ lossAcc,
                             float* __restrict__ out) {
  __shared__ float redc[8], redp[8];
  const int g = blockIdx.x, tid = threadIdx.x;
  const int wave = tid >> 5, lane = tid & 31;
  const float invN = 1.f / (float)NN;
  float cp = 0.f, pp = 0.f;
  for (int mm = tid; mm < NM; mm += 256) {
    const float hp = cnts[g*NM + mm] * invN;
    cp += hp * log2f(hp + 1e-10f);
    const float ap = avgp[g*NM + mm] * invN;
    pp += ap * log2f(ap + 1e-10f);
  }
  #pragma unroll
  for (int off = 16; off; off >>= 1) {
    cp += __shfl_xor(cp, off, 32);
    pp += __shfl_xor(pp, off, 32);
  }
  if (lane == 0) { redc[wave] = cp; redp[wave] = pp; }
  __syncthreads();
  if (tid == 0) {
    float c = 0.f, p = 0.f;
    #pragma unroll
    for (int i = 0; i < 8; ++i) { c += redc[i]; p += redp[i]; }
    out[QOFF + g]     = -c;   // code_perplexity
    out[QOFF + 2 + g] = -p;   // prob_perplexity
    if (g == 0)
      out[QOFF + 4 + (size_t)NN*NG] = lossAcc[0] / ((float)NN * (float)NC);
  }
}

// =====================================================================
extern "C" void kernel_launch(void* const* d_in, const int* in_sizes, int n_in,
                              void* d_out, int out_size, void* d_ws, size_t ws_size,
                              hipStream_t stream) {
  (void)in_sizes; (void)n_in; (void)out_size; (void)ws_size;
  const float* x   = (const float*)d_in[0];
  const float* emb = (const float*)d_in[1];
  const float* gum = (const float*)d_in[2];

  // workspace carve-up
  __bf16* packB   = (__bf16*)d_ws;                       // 524288 bf16 = 1 MB
  float*  sqe     = (float*)((char*)d_ws + 1048576);     // 2048
  float*  avgp    = sqe + 2048;                          // 2048
  float*  cnts    = avgp + 2048;                         // 2048
  float*  lossAcc = cnts + 2048;                         // 1

  float* out      = (float*)d_out;
  int*   outinds  = (int*)((float*)d_out + QOFF + 4);

  gvq_zero<<<(4097 + 255)/256, 256, 0, stream>>>(avgp, 4097);
  gvq_prep<<<NG*64, 256, 0, stream>>>(emb, packB, sqe);
  gvq_main<<<dim3(NN/16, NG), 256, SMEM_BYTES, stream>>>(
      x, emb, gum, packB, sqe, avgp, cnts, lossAcc, out, outinds);
  gvq_finalize<<<NG, 256, 0, stream>>>(cnts, avgp, lossAcc, out);
}
